// VirtualAggr_33818572489172
// MI455X (gfx1250) — compile-verified, matrix operand order checked by reading, not run on hardware
//
#include <hip/hip_runtime.h>
#include <hip/hip_fp16.h>

#define NSEG 8192
#define D    128
#define H    512
#define K1   256     // 2*D
#define MBLK 32      // rows per block in the MLP kernel

typedef __attribute__((ext_vector_type(16))) _Float16 v16h;
typedef __attribute__((ext_vector_type(8)))  float    v8f;

// ---------------------------------------------------------------------------
// Kernel 1: segment sum + counts over sorted segment ids.
// 128 threads/block, thread = one feature column; contiguous row range per
// block; flush with one global atomic per segment boundary per column.
// ---------------------------------------------------------------------------
__global__ void seg_sum_kernel(const float* __restrict__ x,
                               const int*   __restrict__ batch,
                               float* __restrict__ seg_sum,
                               float* __restrict__ counts,
                               int N, int rows_per_block) {
  const int col = threadIdx.x;                       // 0..127
  long r0 = (long)blockIdx.x * rows_per_block;
  long r1 = r0 + rows_per_block; if (r1 > N) r1 = N;
  if (r0 >= r1) return;

  int   cur = batch[r0];
  float acc = 0.0f;
  int   cnt = 0;
  for (long r = r0; r < r1; ++r) {
    const int s = batch[r];
    if (s != cur) {
      atomicAdd(&seg_sum[(long)cur * D + col], acc);
      if (col == 0) atomicAdd(&counts[cur], (float)cnt);
      acc = 0.0f; cnt = 0; cur = s;
    }
    acc += x[r * (long)D + col];
    ++cnt;
  }
  atomicAdd(&seg_sum[(long)cur * D + col], acc);
  if (col == 0) atomicAdd(&counts[cur], (float)cnt);
}

// ---------------------------------------------------------------------------
// Kernel 2: mean + concat with virtual-constraint features, pack to f16.
// X[b][0:128] = seg_sum[b]/max(cnt,1) ; X[b][128:256] = x_virtual_constraints
// ---------------------------------------------------------------------------
__global__ void build_x_kernel(const float* __restrict__ seg,
                               const float* __restrict__ cnt,
                               const float* __restrict__ xvc,
                               _Float16* __restrict__ X) {
  const int b = blockIdx.x;
  const int c = threadIdx.x;                          // 0..127
  const float inv = 1.0f / fmaxf(cnt[b], 1.0f);
  X[(size_t)b * K1 + c]     = (_Float16)(seg[(size_t)b * D + c] * inv);
  X[(size_t)b * K1 + D + c] = (_Float16)(xvc[(size_t)b * D + c]);
}

// ---------------------------------------------------------------------------
// Kernel 3: weight convert + transpose:  Wt[n][k] = (f16) W[k][n]
// ---------------------------------------------------------------------------
__global__ void cvt_transpose_kernel(const float* __restrict__ W,
                                     _Float16* __restrict__ Wt,
                                     int K, int Nn) {
  const int idx = blockIdx.x * blockDim.x + threadIdx.x;
  if (idx >= K * Nn) return;
  const int n = idx / K;
  const int k = idx - n * K;
  Wt[idx] = (_Float16)W[(size_t)k * Nn + n];
}

// ---------------------------------------------------------------------------
// 16-bit 16x32 WMMA fragment load (A layout; B uses the same striping from
// the pre-transposed weights). Lane l: row = l&15, K base = k0 + (l>>4)*8.
// Halves 0..7 at kb, halves 8..15 at kb+16 -> two contiguous b128 loads.
// ---------------------------------------------------------------------------
__device__ __forceinline__ v16h load_frag(const _Float16* base, int ld,
                                          int k0, int lane) {
  const _Float16* p = base + (size_t)(lane & 15) * ld + k0 + ((lane >> 4) << 3);
  v16h r;
  ((uint4*)&r)[0] = ((const uint4*)p)[0];       // K = kb .. kb+7
  ((uint4*)&r)[1] = ((const uint4*)p)[2];       // K = kb+16 .. kb+23
  return r;
}

// ---------------------------------------------------------------------------
// Kernel 4: fused MLP  out = relu(X @ W1 + b1) @ W2 + b2
// 256 threads (8 wave32), 32 rows per block, WMMA f32_16x16x32_f16.
// ---------------------------------------------------------------------------
__global__ void __launch_bounds__(256)
mlp_kernel(const _Float16* __restrict__ X,    // [NSEG][K1]
           const _Float16* __restrict__ W1t,  // [H][K1]   (transposed)
           const float*    __restrict__ b1,   // [H]
           const _Float16* __restrict__ W2t,  // [D][H]    (transposed)
           const float*    __restrict__ b2,   // [D]
           float* __restrict__ out) {         // [NSEG][D]
  __shared__ _Float16 Xs[MBLK * K1];          // 16 KB
  __shared__ _Float16 Hs[MBLK * H];           // 32 KB

  const int tid  = threadIdx.x;
  const int lane = tid & 31;
  const int wid  = tid >> 5;
  const long row0 = (long)blockIdx.x * MBLK;

  // Stage the 32x256 f16 activation tile into LDS (coalesced b128s).
  {
    const uint4* src = (const uint4*)(X + row0 * K1);
    uint4* dst = (uint4*)Xs;
    #pragma unroll
    for (int i = tid; i < (MBLK * K1) / 8; i += 256) dst[i] = src[i];
  }
  __syncthreads();

  // ---- GEMM1: h = relu(X @ W1 + b1), 2x32 tiles of 16x16, K = 256 ----
  for (int t = wid; t < 64; t += 8) {
    const int mt = t >> 5;
    const int nt = t & 31;
    v8f acc = {};
    const _Float16* a_base = Xs + mt * 16 * K1;
    const _Float16* b_base = W1t + (size_t)(nt * 16) * K1;
    #pragma unroll
    for (int ks = 0; ks < K1 / 32; ++ks) {
      v16h a = load_frag(a_base, K1, ks * 32, lane);
      v16h b = load_frag(b_base, K1, ks * 32, lane);
      acc = __builtin_amdgcn_wmma_f32_16x16x32_f16(
          /*neg_a=*/false, a, /*neg_b=*/false, b,
          /*c_mod=*/(short)0, acc, /*reuse_a=*/false, /*reuse_b=*/false);
    }
    // epilogue: C/D layout -> lane 0-15: M=i, N=lane; lane 16-31: M=i+8
    const int n     = nt * 16 + (lane & 15);
    const float bia = b1[n];
    const int mbase = mt * 16 + ((lane >> 4) << 3);
    #pragma unroll
    for (int i = 0; i < 8; ++i) {
      float v = acc[i] + bia;
      v = v > 0.0f ? v : 0.0f;
      Hs[(size_t)(mbase + i) * H + n] = (_Float16)v;
    }
  }
  __syncthreads();

  // ---- GEMM2: out = h @ W2 + b2, 2x8 tiles of 16x16, K = 512 ----
  for (int t = wid; t < 16; t += 8) {
    const int mt = t >> 3;
    const int nt = t & 7;
    v8f acc = {};
    const _Float16* a_base = Hs + mt * 16 * H;
    const _Float16* b_base = W2t + (size_t)(nt * 16) * H;
    #pragma unroll
    for (int ks = 0; ks < H / 32; ++ks) {
      v16h a = load_frag(a_base, H, ks * 32, lane);
      v16h b = load_frag(b_base, H, ks * 32, lane);
      acc = __builtin_amdgcn_wmma_f32_16x16x32_f16(
          false, a, false, b, (short)0, acc, false, false);
    }
    const int n     = nt * 16 + (lane & 15);
    const float bia = b2[n];
    const int mbase = mt * 16 + ((lane >> 4) << 3);
    #pragma unroll
    for (int i = 0; i < 8; ++i) {
      out[(row0 + mbase + i) * D + n] = acc[i] + bia;
    }
  }
}

// ---------------------------------------------------------------------------
extern "C" void kernel_launch(void* const* d_in, const int* in_sizes, int n_in,
                              void* d_out, int out_size, void* d_ws, size_t ws_size,
                              hipStream_t stream) {
  const float* x_variables = (const float*)d_in[0];   // [N,128]
  const float* x_vc        = (const float*)d_in[1];   // [8192,128]
  const int*   batch       = (const int*)  d_in[2];   // [N]
  const float* W1          = (const float*)d_in[3];   // [256,512]
  const float* b1          = (const float*)d_in[4];   // [512]
  const float* W2          = (const float*)d_in[5];   // [512,128]
  const float* b2          = (const float*)d_in[6];   // [128]
  const int N = in_sizes[0] / D;

  // workspace layout
  float*    seg_sum = (float*)d_ws;                       // 8192*128 f32
  float*    counts  = seg_sum + (size_t)NSEG * D;         // 8192    f32
  _Float16* Xh      = (_Float16*)(counts + NSEG);         // 8192*256 f16
  _Float16* W1t     = Xh  + (size_t)NSEG * K1;            // 512*256  f16
  _Float16* W2t     = W1t + (size_t)H * K1;               // 128*512  f16

  hipMemsetAsync(seg_sum, 0, ((size_t)NSEG * D + NSEG) * sizeof(float), stream);

  const int nblk = 4096;
  const int rpb  = (N + nblk - 1) / nblk;
  seg_sum_kernel<<<nblk, D, 0, stream>>>(x_variables, batch, seg_sum, counts, N, rpb);

  build_x_kernel<<<NSEG, D, 0, stream>>>(seg_sum, counts, x_vc, Xh);

  cvt_transpose_kernel<<<(K1 * H + 255) / 256, 256, 0, stream>>>(W1, W1t, K1, H);
  cvt_transpose_kernel<<<(H * D + 255) / 256, 256, 0, stream>>>(W2, W2t, H, D);

  mlp_kernel<<<NSEG / MBLK, 256, 0, stream>>>(Xh, W1t, b1, W2t, b2, (float*)d_out);
}
